// DeepseekV3MoE_60894046323250
// MI455X (gfx1250) — matrix operand order repaired and checked by reading, not compiled, not run on hardware
//
#include <hip/hip_runtime.h>
#include <hip/hip_bf16.h>

// ---------------- problem constants (match reference) ----------------
#define T_TOK   8192        // 4 * 2048 tokens
#define HDIM    2048
#define NEXP    32
#define TOPK    6
#define NGRP    4
#define MDIM    1408
#define SHDIM   2816        // shared experts intermediate
#define NSLOT   (T_TOK * TOPK)   // 49152
#define LDA_STR 40          // padded LDS row stride for A (halfs): 80B, bank-spread

typedef __attribute__((ext_vector_type(16))) _Float16 v16h;
typedef __attribute__((ext_vector_type(8)))  _Float16 v8h;
typedef __attribute__((ext_vector_type(2)))  _Float16 v2h;
typedef __attribute__((ext_vector_type(8)))  float    v8f;

__device__ __forceinline__ float sigmoidf_(float x) {
  return 1.f / (1.f + __expf(-x));
}

// =====================================================================
// LDS tile staging.
// A tile: 64 rows x 32 K, f16, row-major, padded stride LDA_STR halfs.
// B tile: 32 K x 64 N, f16, pre-swizzled into WMMA B-fragment order:
//   element (k, n):  j = n/16, lane = (n%16) + (k>=16 ? 16 : 0), i = k%16
//   lB[(j*32 + lane)*16 + i]
// Wave reads fragment j as two aligned 16-byte ds loads.
// Double-buffered: stage k+32 while computing k (one barrier per step).
// =====================================================================

// stage 64x32 fp32 A rows -> f16 LDS (thread pair per row)
__device__ __forceinline__ void stage_a_f32(const float* __restrict__ rowp,
                                            bool valid,
                                            _Float16* __restrict__ lA,
                                            int tid, int k0) {
  int r   = tid >> 1;
  int seg = (tid & 1) << 4;          // 0 or 16
  _Float16* dst = lA + r * LDA_STR + seg;
  v8h h0, h1;
  if (valid) {
    const float* src = rowp + k0 + seg;
    float4 f0 = *(const float4*)(src);
    float4 f1 = *(const float4*)(src + 4);
    float4 f2 = *(const float4*)(src + 8);
    float4 f3 = *(const float4*)(src + 12);
    h0[0]=(_Float16)f0.x; h0[1]=(_Float16)f0.y; h0[2]=(_Float16)f0.z; h0[3]=(_Float16)f0.w;
    h0[4]=(_Float16)f1.x; h0[5]=(_Float16)f1.y; h0[6]=(_Float16)f1.z; h0[7]=(_Float16)f1.w;
    h1[0]=(_Float16)f2.x; h1[1]=(_Float16)f2.y; h1[2]=(_Float16)f2.z; h1[3]=(_Float16)f2.w;
    h1[4]=(_Float16)f3.x; h1[5]=(_Float16)f3.y; h1[6]=(_Float16)f3.z; h1[7]=(_Float16)f3.w;
  } else {
    #pragma unroll
    for (int i = 0; i < 8; ++i) { h0[i] = (_Float16)0.f; h1[i] = (_Float16)0.f; }
  }
  *(v8h*)dst = h0;
  *(v8h*)(dst + 8) = h1;
}

// stage 64x32 f16 A rows (already f16 source) -> f16 LDS
__device__ __forceinline__ void stage_a_f16(const _Float16* __restrict__ rowp,
                                            bool valid,
                                            _Float16* __restrict__ lA,
                                            int tid, int k0) {
  int r   = tid >> 1;
  int seg = (tid & 1) << 4;
  _Float16* dst = lA + r * LDA_STR + seg;
  v8h h0, h1;
  if (valid) {
    const _Float16* src = rowp + k0 + seg;
    h0 = *(const v8h*)(src);
    h1 = *(const v8h*)(src + 8);
  } else {
    #pragma unroll
    for (int i = 0; i < 8; ++i) { h0[i] = (_Float16)0.f; h1[i] = (_Float16)0.f; }
  }
  *(v8h*)dst = h0;
  *(v8h*)(dst + 8) = h1;
}

// stage 32(K) x 64(N) fp32 weight tile into fragment-swizzled f16 LDS.
// Each thread handles two adjacent K rows x 8 N, writing packed half2
// {k_even, k_odd} -> 8 x ds_store_b32 instead of 16 x b16.
__device__ __forceinline__ void stage_b_f32(const float* __restrict__ wp,
                                            long ld,
                                            _Float16* __restrict__ lB,
                                            int tid) {
  int kp    = tid >> 3;              // 0..15 -> rows k=2kp, 2kp+1
  int ke    = kp << 1;
  int seg   = (tid & 7) << 3;        // n offset: 0..56 step 8
  int j     = seg >> 4;              // n subtile (seg blocks never cross 16)
  int lhi   = (ke & 16) ? 16 : 0;
  int i     = ke & 15;               // even
  int lbase = (seg & 15) + lhi;
  const float* s0 = wp + (long)ke * ld + seg;
  const float* s1 = s0 + ld;
  float4 e0 = *(const float4*)(s0);
  float4 e1 = *(const float4*)(s0 + 4);
  float4 o0 = *(const float4*)(s1);
  float4 o1 = *(const float4*)(s1 + 4);
  float ev[8] = {e0.x,e0.y,e0.z,e0.w, e1.x,e1.y,e1.z,e1.w};
  float ov[8] = {o0.x,o0.y,o0.z,o0.w, o1.x,o1.y,o1.z,o1.w};
  _Float16* dst = lB + (j * 32 + lbase) * 16 + i;
  #pragma unroll
  for (int c = 0; c < 8; ++c) {
    v2h p; p[0] = (_Float16)ev[c]; p[1] = (_Float16)ov[c];
    *(v2h*)(dst + c * 16) = p;
  }
}

// A fragment from padded row-major f16 LDS (16x32, contiguous k = 16h + s)
__device__ __forceinline__ v16h frag_a_lds(const _Float16* __restrict__ lA,
                                           int rowLocal, int lane) {
  int kq = (lane & 16) ? 16 : 0;
  const _Float16* p = lA + rowLocal * LDA_STR;
  v8h lo = *(const v8h*)(p + kq);
  v8h hi = *(const v8h*)(p + kq + 8);
  v16h a;
  #pragma unroll
  for (int i = 0; i < 8; ++i) { a[i] = lo[i]; a[8 + i] = hi[i]; }
  return a;
}

// B fragment j from fragment-swizzled LDS: two aligned 16B loads
__device__ __forceinline__ v16h frag_b_lds(const _Float16* __restrict__ lB,
                                           int j, int lane) {
  const _Float16* p = lB + (j * 32 + lane) * 16;
  v8h lo = *(const v8h*)(p);
  v8h hi = *(const v8h*)(p + 8);
  v16h b;
  #pragma unroll
  for (int i = 0; i < 8; ++i) { b[i] = lo[i]; b[8 + i] = hi[i]; }
  return b;
}

// ---------------- kernel 0: zero counters ----------------
__global__ void moe_init(int* counts, int* cursor) {
  int i = threadIdx.x;
  if (i < NEXP) { counts[i] = 0; cursor[i] = 0; }
}

// ---------------- kernel 1: gating (one wave per token) ----------------
__global__ __launch_bounds__(256) void moe_gate(
    const float* __restrict__ x, const float* __restrict__ gw,
    int* __restrict__ sel, float* __restrict__ wts, int* __restrict__ counts) {
  int wave = threadIdx.x >> 5;
  int lane = threadIdx.x & 31;
  int t = blockIdx.x * 8 + wave;
  if (t >= T_TOK) return;

  const float* xrow = x + (long)t * HDIM;
  float acc = 0.f;
  for (int h0 = 0; h0 < HDIM; h0 += 32) {
    float xv = xrow[h0 + lane];
    #pragma unroll
    for (int j = 0; j < 32; ++j) {
      float xb = __shfl(xv, j, 32);
      acc = fmaf(xb, gw[(long)(h0 + j) * NEXP + lane], acc);
    }
  }
  float score = sigmoidf_(acc);              // lane == expert id

  float gmax = score;
  gmax = fmaxf(gmax, __shfl_xor(gmax, 1, 32));
  gmax = fmaxf(gmax, __shfl_xor(gmax, 2, 32));
  gmax = fmaxf(gmax, __shfl_xor(gmax, 4, 32));

  float gv[NGRP];
  gv[0] = __shfl(gmax, 0, 32);
  gv[1] = __shfl(gmax, 8, 32);
  gv[2] = __shfl(gmax, 16, 32);
  gv[3] = __shfl(gmax, 24, 32);
  int b1 = 0;
  #pragma unroll
  for (int i = 1; i < NGRP; ++i) if (gv[i] > gv[b1]) b1 = i;
  int b2 = (b1 == 0) ? 1 : 0;
  #pragma unroll
  for (int i = 0; i < NGRP; ++i)
    if (i != b1 && gv[i] > gv[b2]) b2 = i;

  int mygrp = lane >> 3;
  float sm = (mygrp == b1 || mygrp == b2) ? score : -__builtin_inff();

  int   selE[TOPK];
  float selW[TOPK];
  #pragma unroll
  for (int k = 0; k < TOPK; ++k) {
    float v = sm; int idx = lane;
    #pragma unroll
    for (int off = 16; off >= 1; off >>= 1) {
      float ov = __shfl_xor(v, off, 32);
      int   oi = __shfl_xor(idx, off, 32);
      if (ov > v || (ov == v && oi < idx)) { v = ov; idx = oi; }
    }
    selE[k] = idx;
    selW[k] = __shfl(score, idx, 32);
    if (lane == idx) sm = -__builtin_inff();
  }
  float wsum = 0.f;
  #pragma unroll
  for (int k = 0; k < TOPK; ++k) wsum += selW[k];

  if (lane == 0) {
    #pragma unroll
    for (int k = 0; k < TOPK; ++k) {
      sel[t * TOPK + k] = selE[k];
      wts[t * TOPK + k] = selW[k] / wsum;
      atomicAdd(&counts[selE[k]], 1);
    }
  }
}

// ---------------- kernel 2: exclusive scan over 32 expert counts ----------------
__global__ void moe_scan(const int* __restrict__ counts, int* __restrict__ offsets) {
  int lane = threadIdx.x;
  int c = counts[lane];
  int s = c;
  #pragma unroll
  for (int off = 1; off < 32; off <<= 1) {
    int o = __shfl_up(s, off, 32);
    if (lane >= off) s += o;
  }
  offsets[lane] = s - c;
}

// ---------------- kernel 3: scatter slots into expert-sorted order ----------------
__global__ __launch_bounds__(256) void moe_scatter(
    const int* __restrict__ sel, const float* __restrict__ wts,
    const int* __restrict__ offsets, int* __restrict__ cursor,
    int* __restrict__ tok_id, float* __restrict__ tok_w) {
  int i = blockIdx.x * 256 + threadIdx.x;
  if (i >= NSLOT) return;
  int e = sel[i];
  int pos = offsets[e] + atomicAdd(&cursor[e], 1);
  tok_id[pos] = i / TOPK;
  tok_w[pos]  = wts[i];
}

// ---------------- kernel 4: grouped gate/up GEMM + SiLU (64x64 tiles) ----------------
__global__ __launch_bounds__(128) void moe_expert_gu(
    const float* __restrict__ x, const float* __restrict__ wg,
    const float* __restrict__ wu, const int* __restrict__ counts,
    const int* __restrict__ offsets, const int* __restrict__ tok_id,
    _Float16* __restrict__ hbuf) {
  __shared__ _Float16 lA[2][64 * LDA_STR];
  __shared__ _Float16 lBg[2][2048];
  __shared__ _Float16 lBu[2][2048];

  int e    = blockIdx.y;
  int n0   = blockIdx.x * 64;
  int tid  = threadIdx.x;
  int wave = tid >> 5;
  int lane = tid & 31;
  int cnt  = counts[e];
  int base = offsets[e];
  const float* wgE = wg + (long)e * HDIM * MDIM;
  const float* wuE = wu + (long)e * HDIM * MDIM;

  for (int r0 = 0; r0 < cnt; r0 += 64) {          // uniform across block
    int  rA = tid >> 1;
    bool vA = (r0 + rA) < cnt;
    const float* rowpA = x + (long)tok_id[base + r0 + (vA ? rA : 0)] * HDIM;

    v8f accg[4] = {};
    v8f accu[4] = {};

    __syncthreads();                              // prev block's readers done
    stage_a_f32(rowpA, vA, lA[0], tid, 0);
    stage_b_f32(wgE + n0, MDIM, lBg[0], tid);
    stage_b_f32(wuE + n0, MDIM, lBu[0], tid);

    int cur = 0;
    for (int k0 = 0; k0 < HDIM; k0 += 32) {
      __syncthreads();                            // buf[cur] staged; buf[nxt] free
      int nxt = cur ^ 1;
      if (k0 + 32 < HDIM) {
        stage_a_f32(rowpA, vA, lA[nxt], tid, k0 + 32);
        stage_b_f32(wgE + (long)(k0 + 32) * MDIM + n0, MDIM, lBg[nxt], tid);
        stage_b_f32(wuE + (long)(k0 + 32) * MDIM + n0, MDIM, lBu[nxt], tid);
        if (k0 + 64 < HDIM && tid < 32) {
          __builtin_prefetch(wgE + (long)(k0 + 64 + tid) * MDIM + n0, 0, 1);
          __builtin_prefetch(wuE + (long)(k0 + 64 + tid) * MDIM + n0, 0, 1);
        }
      }
      v16h a = frag_a_lds(lA[cur], wave * 16 + (lane & 15), lane);
      #pragma unroll
      for (int j = 0; j < 4; ++j) {
        v16h bg = frag_b_lds(lBg[cur], j, lane);
        v16h bu = frag_b_lds(lBu[cur], j, lane);
        accg[j] = __builtin_amdgcn_wmma_f32_16x16x32_f16(false, a, false, bg,
                                                         (short)0, accg[j],
                                                         false, false);
        accu[j] = __builtin_amdgcn_wmma_f32_16x16x32_f16(false, a, false, bu,
                                                         (short)0, accu[j],
                                                         false, false);
      }
      cur = nxt;
    }

    int col = lane & 15;
    int rh  = (lane & 16) ? 8 : 0;
    #pragma unroll
    for (int r = 0; r < 8; ++r) {
      int row = r0 + wave * 16 + rh + r;
      if (row < cnt) {
        #pragma unroll
        for (int j = 0; j < 4; ++j) {
          float g = accg[j][r], u = accu[j][r];
          float h = (g * sigmoidf_(g)) * u;       // silu(g) * u
          hbuf[(long)(base + row) * MDIM + n0 + j * 16 + col] = (_Float16)h;
        }
      }
    }
  }
}

// ---------------- kernel 5: grouped down GEMM, scale, accumulate ----------------
__global__ __launch_bounds__(128) void moe_expert_down(
    const _Float16* __restrict__ hbuf, const float* __restrict__ wd,
    const int* __restrict__ counts, const int* __restrict__ offsets,
    const int* __restrict__ tok_id, const float* __restrict__ tok_w,
    float* __restrict__ out) {
  __shared__ _Float16 lA[2][64 * LDA_STR];
  __shared__ _Float16 lB[2][2048];

  int e    = blockIdx.y;
  int h0   = blockIdx.x * 64;
  int tid  = threadIdx.x;
  int wave = tid >> 5;
  int lane = tid & 31;
  int cnt  = counts[e];
  int base = offsets[e];
  const float* wdE = wd + (long)e * MDIM * HDIM;

  for (int r0 = 0; r0 < cnt; r0 += 64) {
    int  rA = tid >> 1;
    bool vA = (r0 + rA) < cnt;
    const _Float16* rowpA = hbuf + (long)(base + r0 + (vA ? rA : 0)) * MDIM;

    v8f acc[4] = {};

    __syncthreads();
    stage_a_f16(rowpA, vA, lA[0], tid, 0);
    stage_b_f32(wdE + h0, HDIM, lB[0], tid);

    int cur = 0;
    for (int k0 = 0; k0 < MDIM; k0 += 32) {
      __syncthreads();
      int nxt = cur ^ 1;
      if (k0 + 32 < MDIM) {
        stage_a_f16(rowpA, vA, lA[nxt], tid, k0 + 32);
        stage_b_f32(wdE + (long)(k0 + 32) * HDIM + h0, HDIM, lB[nxt], tid);
        if (k0 + 64 < MDIM && tid < 32)
          __builtin_prefetch(wdE + (long)(k0 + 64 + tid) * HDIM + h0, 0, 1);
      }
      v16h a = frag_a_lds(lA[cur], wave * 16 + (lane & 15), lane);
      #pragma unroll
      for (int j = 0; j < 4; ++j) {
        v16h b = frag_b_lds(lB[cur], j, lane);
        acc[j] = __builtin_amdgcn_wmma_f32_16x16x32_f16(false, a, false, b,
                                                        (short)0, acc[j],
                                                        false, false);
      }
      cur = nxt;
    }

    int col = lane & 15;
    int rh  = (lane & 16) ? 8 : 0;
    #pragma unroll
    for (int r = 0; r < 8; ++r) {
      int row = r0 + wave * 16 + rh + r;
      if (row < cnt) {
        int slot = base + row;
        int t    = tok_id[slot];
        float w  = tok_w[slot];
        #pragma unroll
        for (int j = 0; j < 4; ++j)
          atomicAdd(&out[(long)t * HDIM + h0 + j * 16 + col], acc[j][r] * w);
      }
    }
  }
}

// ---------------- kernel 6: shared expert gate/up + SiLU (64x64 tiles) ----------------
__global__ __launch_bounds__(128) void shared_gu(
    const float* __restrict__ x, const float* __restrict__ swg,
    const float* __restrict__ swu, _Float16* __restrict__ sbuf) {
  __shared__ _Float16 lA[2][64 * LDA_STR];
  __shared__ _Float16 lBg[2][2048];
  __shared__ _Float16 lBu[2][2048];

  int n0   = blockIdx.x * 64;        // SH/64 tiles
  int row0 = blockIdx.y * 64;        // T/64 tiles
  int tid  = threadIdx.x;
  int wave = tid >> 5;
  int lane = tid & 31;
  const float* rowpA = x + (long)(row0 + (tid >> 1)) * HDIM;

  v8f accg[4] = {};
  v8f accu[4] = {};

  stage_a_f32(rowpA, true, lA[0], tid, 0);
  stage_b_f32(swg + n0, SHDIM, lBg[0], tid);
  stage_b_f32(swu + n0, SHDIM, lBu[0], tid);

  int cur = 0;
  for (int k0 = 0; k0 < HDIM; k0 += 32) {
    __syncthreads();
    int nxt = cur ^ 1;
    if (k0 + 32 < HDIM) {
      stage_a_f32(rowpA, true, lA[nxt], tid, k0 + 32);
      stage_b_f32(swg + (long)(k0 + 32) * SHDIM + n0, SHDIM, lBg[nxt], tid);
      stage_b_f32(swu + (long)(k0 + 32) * SHDIM + n0, SHDIM, lBu[nxt], tid);
      if (k0 + 64 < HDIM && tid < 32) {
        __builtin_prefetch(swg + (long)(k0 + 64 + tid) * SHDIM + n0, 0, 1);
        __builtin_prefetch(swu + (long)(k0 + 64 + tid) * SHDIM + n0, 0, 1);
      }
    }
    v16h a = frag_a_lds(lA[cur], wave * 16 + (lane & 15), lane);
    #pragma unroll
    for (int j = 0; j < 4; ++j) {
      v16h bg = frag_b_lds(lBg[cur], j, lane);
      v16h bu = frag_b_lds(lBu[cur], j, lane);
      accg[j] = __builtin_amdgcn_wmma_f32_16x16x32_f16(false, a, false, bg,
                                                       (short)0, accg[j],
                                                       false, false);
      accu[j] = __builtin_amdgcn_wmma_f32_16x16x32_f16(false, a, false, bu,
                                                       (short)0, accu[j],
                                                       false, false);
    }
    cur = nxt;
  }

  int col = lane & 15;
  int rh  = (lane & 16) ? 8 : 0;
  #pragma unroll
  for (int r = 0; r < 8; ++r) {
    int row = row0 + wave * 16 + rh + r;
    #pragma unroll
    for (int j = 0; j < 4; ++j) {
      float g = accg[j][r], u = accu[j][r];
      float h = (g * sigmoidf_(g)) * u;
      sbuf[(long)row * SHDIM + n0 + j * 16 + col] = (_Float16)h;
    }
  }
}

// ---------------- kernel 7: shared down GEMM (initializes d_out) ----------------
__global__ __launch_bounds__(128) void shared_down(
    const _Float16* __restrict__ sbuf, const float* __restrict__ swd,
    float* __restrict__ out) {
  __shared__ _Float16 lA[2][64 * LDA_STR];
  __shared__ _Float16 lB[2][2048];

  int h0   = blockIdx.x * 64;        // H/64 tiles
  int row0 = blockIdx.y * 64;        // T/64 tiles
  int tid  = threadIdx.x;
  int wave = tid >> 5;
  int lane = tid & 31;
  const _Float16* rowpA = sbuf + (long)(row0 + (tid >> 1)) * SHDIM;

  v8f acc[4] = {};

  stage_a_f16(rowpA, true, lA[0], tid, 0);
  stage_b_f32(swd + h0, HDIM, lB[0], tid);

  int cur = 0;
  for (int k0 = 0; k0 < SHDIM; k0 += 32) {
    __syncthreads();
    int nxt = cur ^ 1;
    if (k0 + 32 < SHDIM) {
      stage_a_f16(rowpA, true, lA[nxt], tid, k0 + 32);
      stage_b_f32(swd + (long)(k0 + 32) * HDIM + h0, HDIM, lB[nxt], tid);
      if (k0 + 64 < SHDIM && tid < 32)
        __builtin_prefetch(swd + (long)(k0 + 64 + tid) * HDIM + h0, 0, 1);
    }
    v16h a = frag_a_lds(lA[cur], wave * 16 + (lane & 15), lane);
    #pragma unroll
    for (int j = 0; j < 4; ++j) {
      v16h b = frag_b_lds(lB[cur], j, lane);
      acc[j] = __builtin_amdgcn_wmma_f32_16x16x32_f16(false, a, false, b,
                                                      (short)0, acc[j],
                                                      false, false);
    }
    cur = nxt;
  }

  int col = lane & 15;
  int rh  = (lane & 16) ? 8 : 0;
  #pragma unroll
  for (int r = 0; r < 8; ++r) {
    int row = row0 + wave * 16 + rh + r;
    #pragma unroll
    for (int j = 0; j < 4; ++j)
      out[(long)row * HDIM + h0 + j * 16 + col] = acc[j][r];
  }
}

// ---------------- host side ----------------
extern "C" void kernel_launch(void* const* d_in, const int* in_sizes, int n_in,
                              void* d_out, int out_size, void* d_ws, size_t ws_size,
                              hipStream_t stream) {
  const float* x   = (const float*)d_in[0];
  const float* gw  = (const float*)d_in[1];
  const float* wg  = (const float*)d_in[2];
  const float* wu  = (const float*)d_in[3];
  const float* wd  = (const float*)d_in[4];
  const float* swg = (const float*)d_in[5];
  const float* swu = (const float*)d_in[6];
  const float* swd = (const float*)d_in[7];
  float* out = (float*)d_out;

  char* ws = (char*)d_ws;
  int*   counts  = (int*)ws;          ws += 256;
  int*   cursor  = (int*)ws;          ws += 256;
  int*   offsets = (int*)ws;          ws += 256;
  int*   sel     = (int*)ws;          ws += (size_t)NSLOT * sizeof(int);
  float* wts     = (float*)ws;        ws += (size_t)NSLOT * sizeof(float);
  int*   tok_id  = (int*)ws;          ws += (size_t)NSLOT * sizeof(int);
  float* tok_w   = (float*)ws;        ws += (size_t)NSLOT * sizeof(float);
  _Float16* hbuf = (_Float16*)ws;     ws += (size_t)NSLOT * MDIM * sizeof(_Float16);
  _Float16* sbuf = (_Float16*)ws;

  moe_init   <<<1, 64, 0, stream>>>(counts, cursor);
  moe_gate   <<<T_TOK / 8, 256, 0, stream>>>(x, gw, sel, wts, counts);
  moe_scan   <<<1, 32, 0, stream>>>(counts, offsets);
  moe_scatter<<<NSLOT / 256, 256, 0, stream>>>(sel, wts, offsets, cursor,
                                               tok_id, tok_w);
  // shared expert path (shared_down fully writes d_out)
  shared_gu  <<<dim3(SHDIM / 64, T_TOK / 64), 128, 0, stream>>>(x, swg, swu, sbuf);
  shared_down<<<dim3(HDIM / 64, T_TOK / 64), 128, 0, stream>>>(sbuf, swd, out);
  // routed experts (expert-major grid keeps each expert's weights hot in L2)
  moe_expert_gu  <<<dim3(MDIM / 64, NEXP), 128, 0, stream>>>(
      x, wg, wu, counts, offsets, tok_id, hbuf);
  moe_expert_down<<<dim3(HDIM / 64, NEXP), 128, 0, stream>>>(
      hbuf, wd, counts, offsets, tok_id, tok_w, out);
}
